// NATransformerLayer_68685116998001
// MI455X (gfx1250) — matrix operand order repaired and checked by reading, not compiled, not run on hardware
//
#include <hip/hip_runtime.h>
#include <hip/hip_bf16.h>

typedef _Float16 v16h __attribute__((ext_vector_type(16)));
typedef _Float16 v8h  __attribute__((ext_vector_type(8)));
typedef _Float16 v4h  __attribute__((ext_vector_type(4)));
typedef float    v8f  __attribute__((ext_vector_type(8)));
typedef unsigned u32x4 __attribute__((ext_vector_type(4)));
typedef int      i32x8 __attribute__((ext_vector_type(8)));
typedef int      i32x4 __attribute__((ext_vector_type(4)));

#define TOK   18432      // 2*96*96
#define CDIM  128
#define HSZ   96
#define NHEAD 4
#define HD    32
#define KS    7

union FragH { v16h v; v8h h[2]; };

// ---------------- f32 -> f16 weight conversion ----------------
__global__ void nat_cvt_f16(const float* __restrict__ src, _Float16* __restrict__ dst, int n) {
    int i = blockIdx.x * blockDim.x + threadIdx.x;
    if (i < n) dst[i] = (_Float16)src[i];
}

// ---------------- LayerNorm (one wave32 per row of 128) ----------------
__global__ void nat_layernorm(const float* __restrict__ x, const float* __restrict__ g,
                              const float* __restrict__ b, _Float16* __restrict__ y, int rows) {
    int wave = (blockIdx.x * blockDim.x + threadIdx.x) >> 5;
    int lane = threadIdx.x & 31;
    if (wave >= rows) return;
    const float4* xr = (const float4*)(x + (size_t)wave * CDIM);
    float4 v = xr[lane];
    float s = v.x + v.y + v.z + v.w;
    #pragma unroll
    for (int o = 16; o; o >>= 1) s += __shfl_xor(s, o, 32);
    float mean = s * (1.0f / 128.0f);
    float dx = v.x - mean, dy = v.y - mean, dz = v.z - mean, dw = v.w - mean;
    float s2 = dx * dx + dy * dy + dz * dz + dw * dw;
    #pragma unroll
    for (int o = 16; o; o >>= 1) s2 += __shfl_xor(s2, o, 32);
    float inv = rsqrtf(s2 * (1.0f / 128.0f) + 1e-6f);
    float4 gg = ((const float4*)g)[lane];
    float4 bb = ((const float4*)b)[lane];
    v4h out;
    out[0] = (_Float16)(dx * inv * gg.x + bb.x);
    out[1] = (_Float16)(dy * inv * gg.y + bb.y);
    out[2] = (_Float16)(dz * inv * gg.z + bb.z);
    out[3] = (_Float16)(dw * inv * gg.w + bb.w);
    ((v4h*)(y + (size_t)wave * CDIM))[lane] = out;
}

// ---- WMMA GEMM: Y[M,N] = A[M,128] @ W[N,128]^T (+bias)(*scale?)(+residual?) ----
// Block = 8 waves sharing one 16-col n-tile; W tile (16x128 f16 = 4KB) DMA'd into LDS
// by the Tensor Data Mover (one tensor_load_to_lds per block), B fragments then read
// via ds_load. Each wave computes a 64x16 tile (4 accumulators, 16x v_wmma).
template <bool HAS_RES, bool HAS_SCALE>
__global__ void nat_wmma_gemm(const _Float16* __restrict__ A, const _Float16* __restrict__ W,
                              const float* __restrict__ bias, const float* __restrict__ residual,
                              const float* __restrict__ scale, float* __restrict__ Y,
                              int M, int N) {
    __shared__ __align__(16) _Float16 wtile[16 * CDIM];   // 4 KB

    int lane = threadIdx.x & 31;
    int n0 = blockIdx.x << 4;                              // n-tile (shared by block)
    int m0 = (blockIdx.y << 9) + ((threadIdx.x >> 5) << 6);// 512 rows/block, 64 rows/wave

#if __has_builtin(__builtin_amdgcn_tensor_load_to_lds)
    if (threadIdx.x < 32) {
        // ---- Tensor DMA descriptor (CDNA5 ISA) ----
        unsigned long long ga = (unsigned long long)(W + (size_t)n0 * CDIM);
        unsigned lds_off = (unsigned)(unsigned long long)(const void*)wtile;
        u32x4 g0;
        g0[0] = 1u;                                        // count=1, user desc
        g0[1] = lds_off;                                   // lds_addr
        g0[2] = (unsigned)ga;                              // global_addr[31:0]
        g0[3] = (unsigned)(ga >> 32) | 0x80000000u;        // addr[56:32] | type=2 (bits 127:126)
        i32x8 g1;
        const unsigned tdim0 = 256u;                       // bytes per row (data_size = 1B)
        g1[0] = 0;                                         // wg_mask=0, data_size=0 (1B), no flags
        g1[1] = (int)(tdim0 << 16);                        // tensor_dim0[15:0] << 16
        g1[2] = (int)(16u << 16);                          // tensor_dim0[31:16]=0 | tensor_dim1=16
        g1[3] = (int)(tdim0 << 16);                        // tensor_dim1[31:16]=0 | tile_dim0=256
        g1[4] = 16;                                        // tile_dim1=16, tile_dim2=0
        g1[5] = (int)tdim0;                                // tensor_dim0_stride[31:0]
        g1[6] = 0;                                         // stride0[47:32] | stride1[15:0]
        g1[7] = 0;                                         // stride1[47:16]
        i32x4 gz4 = {0, 0, 0, 0};                          // groups 2/3 unused (2D tensor)
        i32x8 gz8 = {0, 0, 0, 0, 0, 0, 0, 0};
        __builtin_amdgcn_tensor_load_to_lds(g0, g1, gz4, gz4, gz8, 0);
#if __has_builtin(__builtin_amdgcn_s_wait_tensorcnt)
        __builtin_amdgcn_s_wait_tensorcnt(0);
#else
        asm volatile("s_wait_tensorcnt 0x0" ::: "memory");
#endif
    }
#else
    // fallback: cooperative load, 256 threads x 16B = 4KB
    ((v8h*)wtile)[threadIdx.x] = ((const v8h*)(W + (size_t)n0 * CDIM))[threadIdx.x];
#endif
    __syncthreads();

    // A fragment: lane l holds row (l&15); K groups (l>>4)*8 and +16 within each 32-chunk
    int agrp = (lane >> 4) << 3;
    const _Float16* abase = A + (size_t)(m0 + (lane & 15)) * CDIM;
    // B fragment from LDS: lane l holds column (l&15); K groups (l>>4)*8 and +16 within each 32-chunk
    const _Float16* wl = wtile + (lane & 15) * CDIM + ((lane >> 4) << 3);
    int ncol = n0 + (lane & 15);

    v8f acc[4] = {};
    #pragma unroll
    for (int kk = 0; kk < 4; ++kk) {
        int kb = kk * 32;
        FragH bf;
        bf.h[0] = *(const v8h*)(wl + kb);
        bf.h[1] = *(const v8h*)(wl + kb + 16);
        #pragma unroll
        for (int sub = 0; sub < 4; ++sub) {
            const _Float16* ap = abase + (size_t)(sub * 16) * CDIM + kb + agrp;
            FragH a;
            a.h[0] = *(const v8h*)(ap);
            a.h[1] = *(const v8h*)(ap + 16);
            acc[sub] = __builtin_amdgcn_wmma_f32_16x16x32_f16(false, a.v, false, bf.v,
                                                              (short)0, acc[sub], false, false);
        }
    }

    float bn = bias[ncol];
    float sc = HAS_SCALE ? scale[ncol] : 1.0f;
    int rbase = (lane >> 4) << 3;
    #pragma unroll
    for (int sub = 0; sub < 4; ++sub) {
        #pragma unroll
        for (int r = 0; r < 8; ++r) {
            size_t idx = (size_t)(m0 + sub * 16 + rbase + r) * N + ncol;
            float val = (acc[sub][r] + bn) * sc;
            if (HAS_RES) val += residual[idx];
            Y[idx] = val;
        }
    }
}

// ---------------- Neighborhood attention (thread per query x head, online softmax) --------
__global__ void nat_natten(const float* __restrict__ qkv, const float* __restrict__ rpb,
                           _Float16* __restrict__ out) {
    int idx = blockIdx.x * blockDim.x + threadIdx.x;
    const int HW = HSZ * HSZ;
    if (idx >= 2 * NHEAD * HW) return;
    int j    = idx % HSZ;
    int i    = (idx / HSZ) % HSZ;
    int head = (idx / HW) % NHEAD;
    int b    = idx / (HW * NHEAD);

    const float qscale = 0.17677669529663687f; // 32^-0.5
    const float* qp = qkv + ((size_t)((b * HSZ + i) * HSZ + j)) * 384 + head * HD;
    float4 q[8];
    #pragma unroll
    for (int d = 0; d < 8; ++d) {
        q[d] = ((const float4*)qp)[d];
        q[d].x *= qscale; q[d].y *= qscale; q[d].z *= qscale; q[d].w *= qscale;
    }

    int si = i - KS / 2; si = si < 0 ? 0 : (si > HSZ - KS ? HSZ - KS : si);
    int sj = j - KS / 2; sj = sj < 0 ? 0 : (sj > HSZ - KS ? HSZ - KS : sj);
    const float* rb = rpb + head * (2 * KS - 1) * (2 * KS - 1)
                          + (si - i + KS - 1) * (2 * KS - 1) + (sj - j + KS - 1);

    float mx = -1e30f, l = 0.0f;
    float4 o[8];
    #pragma unroll
    for (int d = 0; d < 8; ++d) o[d] = make_float4(0.f, 0.f, 0.f, 0.f);

    for (int a = 0; a < KS; ++a) {
        for (int c = 0; c < KS; ++c) {
            const float* kp = qkv + ((size_t)((b * HSZ + si + a) * HSZ + sj + c)) * 384
                                  + CDIM + head * HD;
            float s = 0.f;
            #pragma unroll
            for (int d = 0; d < 8; ++d) {
                float4 kv = ((const float4*)kp)[d];
                s += q[d].x * kv.x + q[d].y * kv.y + q[d].z * kv.z + q[d].w * kv.w;
            }
            s += rb[a * (2 * KS - 1) + c];
            float mnew = fmaxf(mx, s);
            float corr = __expf(mx - mnew);
            float p    = __expf(s - mnew);
            l = l * corr + p;
            const float* vp = kp + CDIM;
            #pragma unroll
            for (int d = 0; d < 8; ++d) {
                float4 vv = ((const float4*)vp)[d];
                o[d].x = o[d].x * corr + p * vv.x;
                o[d].y = o[d].y * corr + p * vv.y;
                o[d].z = o[d].z * corr + p * vv.z;
                o[d].w = o[d].w * corr + p * vv.w;
            }
            mx = mnew;
        }
    }
    float linv = 1.0f / l;
    _Float16* op = out + ((size_t)((b * HSZ + i) * HSZ + j)) * CDIM + head * HD;
    #pragma unroll
    for (int d = 0; d < 8; ++d) {
        v4h h;
        h[0] = (_Float16)(o[d].x * linv);
        h[1] = (_Float16)(o[d].y * linv);
        h[2] = (_Float16)(o[d].z * linv);
        h[3] = (_Float16)(o[d].w * linv);
        ((v4h*)op)[d] = h;
    }
}

// ---------------- simple gate: g = h[:, :128] * h[:, 128:256] -> f16 ----------------
__global__ void nat_gate(const float* __restrict__ h, _Float16* __restrict__ g, int n) {
    int idx = blockIdx.x * blockDim.x + threadIdx.x;
    if (idx >= n) return;
    int m = idx >> 7, c = idx & 127;
    g[idx] = (_Float16)(h[(size_t)m * 256 + c] * h[(size_t)m * 256 + 128 + c]);
}

extern "C" void kernel_launch(void* const* d_in, const int* in_sizes, int n_in,
                              void* d_out, int out_size, void* d_ws, size_t ws_size,
                              hipStream_t stream) {
    const float* x      = (const float*)d_in[0];
    const float* qkv_w  = (const float*)d_in[1];
    const float* qkv_b  = (const float*)d_in[2];
    const float* proj_w = (const float*)d_in[3];
    const float* proj_b = (const float*)d_in[4];
    const float* rpb    = (const float*)d_in[5];
    const float* ln1_g  = (const float*)d_in[6];
    const float* ln1_b  = (const float*)d_in[7];
    const float* ln2_g  = (const float*)d_in[8];
    const float* ln2_b  = (const float*)d_in[9];
    const float* w4     = (const float*)d_in[10];
    const float* b4     = (const float*)d_in[11];
    const float* w5     = (const float*)d_in[12];
    const float* b5     = (const float*)d_in[13];
    const float* beta   = (const float*)d_in[14];
    float* out = (float*)d_out;

    char* ws = (char*)d_ws;
    const int M = TOK;
    // workspace layout (bytes)
    float*     qkv    = (float*)(ws);                                  // [M,384] f32
    _Float16*  attn_h = (_Float16*)(ws + 28311552);                    // [M,128] f16 (reused as g_h)
    float*     x1     = (float*)(ws + 33030144);                       // [M,128] f32
    float*     hbuf   = (float*)(ws + 42467328);                       // [M,256] f32
    _Float16*  xh     = (_Float16*)(ws + 61341696);                    // [M,128] f16
    _Float16*  wq_h   = (_Float16*)(ws + 66060288);                    // 384*128
    _Float16*  wp_h   = (_Float16*)(ws + 66060288 + 98304);            // 128*128
    _Float16*  w4_h   = (_Float16*)(ws + 66060288 + 98304 + 32768);    // 256*128
    _Float16*  w5_h   = (_Float16*)(ws + 66060288 + 98304 + 32768 + 65536); // 128*128

    // weight conversions
    nat_cvt_f16<<<(384 * 128 + 255) / 256, 256, 0, stream>>>(qkv_w, wq_h, 384 * 128);
    nat_cvt_f16<<<(128 * 128 + 255) / 256, 256, 0, stream>>>(proj_w, wp_h, 128 * 128);
    nat_cvt_f16<<<(256 * 128 + 255) / 256, 256, 0, stream>>>(w4, w4_h, 256 * 128);
    nat_cvt_f16<<<(128 * 128 + 255) / 256, 256, 0, stream>>>(w5, w5_h, 128 * 128);

    // 1) LN1 -> xh (f16)
    nat_layernorm<<<M / 8, 256, 0, stream>>>(x, ln1_g, ln1_b, xh, M);

    // 2) qkv = xh @ qkv_w^T + qkv_b   [M,384]
    nat_wmma_gemm<false, false><<<dim3(384 / 16, M / 512), 256, 0, stream>>>(
        xh, wq_h, qkv_b, nullptr, nullptr, qkv, M, 384);

    // 3) neighborhood attention -> attn_h (f16)
    nat_natten<<<(2 * NHEAD * HSZ * HSZ + 255) / 256, 256, 0, stream>>>(qkv, rpb, attn_h);

    // 4) x1 = x + attn_h @ proj_w^T + proj_b
    nat_wmma_gemm<true, false><<<dim3(128 / 16, M / 512), 256, 0, stream>>>(
        attn_h, wp_h, proj_b, x, nullptr, x1, M, 128);

    // 5) LN2 -> xh (f16, reuse)
    nat_layernorm<<<M / 8, 256, 0, stream>>>(x1, ln2_g, ln2_b, xh, M);

    // 6) hbuf = xh @ w4^T + b4   [M,256]
    nat_wmma_gemm<false, false><<<dim3(256 / 16, M / 512), 256, 0, stream>>>(
        xh, w4_h, b4, nullptr, nullptr, hbuf, M, 256);

    // 7) gate -> g_h (reuse attn_h buffer)
    nat_gate<<<(M * 128 + 255) / 256, 256, 0, stream>>>(hbuf, attn_h, M * 128);

    // 8) out = x1 + beta * (g_h @ w5^T + b5)
    nat_wmma_gemm<true, true><<<dim3(128 / 16, M / 512), 256, 0, stream>>>(
        attn_h, w5_h, b5, x1, beta, out, M, 128);
}
